// MetalSaltGNN_Energy_67654324847093
// MI455X (gfx1250) — matrix operand order repaired and checked by reading, not compiled, not run on hardware
//
#include <hip/hip_runtime.h>

#define HID 128
#define NODE_IN 92
#define EDGE_IN 16
#define LAT_IN 9
#define BN_EPS 1e-5f

typedef __attribute__((ext_vector_type(16))) __bf16 v16bf;
typedef __attribute__((ext_vector_type(8)))  float  v8f;

union AB16 { v16bf v; uint4 q[2]; };

__device__ __forceinline__ v8f wmma_bf16(v16bf a, v16bf b, v8f c) {
  return __builtin_amdgcn_wmma_f32_16x16x32_bf16(false, a, false, b, (short)0, c, false, false);
}

// ---- A-tile from a bf16 row (global e or LDS hidden): pure b128 loads ----
// elements 0..7  <- row[k0 + 8*hi .. +7]      (VGPRs 0-3)
// elements 8..15 <- row[k0 + 16 + 8*hi .. +7] (VGPRs 4-7)
template <typename PT>
__device__ __forceinline__ v16bf load_a_bf16_row(PT row, int hi, int k0) {
  AB16 u;
  u.q[0] = *(const uint4*)(row + k0 + hi * 8);
  u.q[1] = *(const uint4*)(row + k0 + 16 + hi * 8);
  return u.v;
}

// ---- A-tile from f32 row, full 32-K validity (vectorizable loads + cvt_pk) ----
__device__ __forceinline__ v16bf load_a_f32_row(const float* row, int hi, int k0) {
  v16bf a;
  const float* p0 = row + k0 + hi * 8;
  const float* p1 = row + k0 + 16 + hi * 8;
#pragma unroll
  for (int j = 0; j < 8; ++j) {
    a[j]     = (__bf16)p0[j];
    a[8 + j] = (__bf16)p1[j];
  }
  return a;
}

// ---- A-tile = (x + aggr) rows, f32 ----
__device__ __forceinline__ v16bf load_a_sum(const float* r0, const float* r1, int hi, int k0) {
  v16bf a;
  const int o0 = k0 + hi * 8, o1 = k0 + 16 + hi * 8;
#pragma unroll
  for (int j = 0; j < 8; ++j) {
    a[j]     = (__bf16)(r0[o0 + j] + r1[o0 + j]);
    a[8 + j] = (__bf16)(r0[o1 + j] + r1[o1 + j]);
  }
  return a;
}

// ---- edge_attr A tile: K=16 real (first half always valid, second half zero) ----
__device__ __forceinline__ v16bf load_a_ea(const float* row, int hi) {
  v16bf a;
  const float* p0 = row + hi * 8;
#pragma unroll
  for (int j = 0; j < 8; ++j) {
    a[j]     = (__bf16)p0[j];
    a[8 + j] = (__bf16)0.f;
  }
  return a;
}

// ---- lattice A tile: K=9 real ----
__device__ __forceinline__ v16bf load_a_lat(const float* row, int hi) {
  v16bf a;
#pragma unroll
  for (int j = 0; j < 8; ++j) {
    int k = hi * 8 + j;
    a[j]     = (__bf16)((k < LAT_IN) ? row[k] : 0.f);
    a[8 + j] = (__bf16)0.f;
  }
  return a;
}

// ---- staged B fragment: 32 contiguous bytes per lane -> 2x b128, no cvt ----
__device__ __forceinline__ v16bf load_b_staged(const __bf16* wst, int kt, int nt, int lane) {
  const __bf16* p = wst + (((size_t)(kt * 8 + nt) * 32 + lane) << 4);
  AB16 u;
  u.q[0] = ((const uint4*)p)[0];
  u.q[1] = ((const uint4*)p)[1];
  return u.v;
}

// ================= staging / utility kernels =================

// Stage f32 W[K,Nc] (row-major) into bf16 WMMA-B layout, padded to KP x NP (NP==128 here).
__global__ void k_stage_b(const float* __restrict__ w, __bf16* __restrict__ out,
                          int K, int Nc, int KP, int NP) {
  int i = blockIdx.x * blockDim.x + threadIdx.x;
  int total = KP * NP;
  if (i >= total) return;
  int tile = i >> 9, within = i & 511;
  int lane = within >> 4, elem = within & 15;
  int ntn = NP >> 4;
  int kt = tile / ntn, nt = tile % ntn;
  int hi = lane >> 4, lo = lane & 15;
  int k = kt * 32 + hi * 8 + (elem & 7) + (elem >> 3) * 16;   // elem<8 -> K 8*hi+elem, elem>=8 -> K 16+8*hi+(elem-8)
  int n = nt * 16 + lo;
  float val = (k < K && n < Nc) ? w[(size_t)k * Nc + n] : 0.f;
  out[i] = (__bf16)val;
}

__global__ void k_pad_vec(const float* __restrict__ in, float* __restrict__ out, int n) {
  int i = threadIdx.x;
  if (i < HID) out[i] = (i < n) ? in[i] : 0.f;
}

__global__ void k_pad_x(const float* __restrict__ x, float* __restrict__ xp, int n_elem) {
  int i = blockIdx.x * blockDim.x + threadIdx.x;
  if (i >= n_elem) return;
  int n = i >> 7, c = i & 127;
  xp[i] = (c < NODE_IN) ? x[n * NODE_IN + c] : 0.f;
}

__global__ void k_zero(float* __restrict__ p, int n) {
  int i = blockIdx.x * blockDim.x + threadIdx.x;
  if (i < n) p[i] = 0.f;
}

// ================= main kernels =================

// e = W1 @ relu(W0 @ edge_attr + b0) + b1, stored bf16 [E,128]
__global__ __launch_bounds__(128) void k_edge_encoder(
    const float* __restrict__ ea, const __bf16* __restrict__ w0s, const float* __restrict__ b0,
    const __bf16* __restrict__ w1s, const float* __restrict__ b1,
    __bf16* __restrict__ e_out, int ntiles) {
  __shared__ __bf16 lds[4][16][160];
  const int lane = threadIdx.x & 31;
  const int wv = threadIdx.x >> 5;
  const int tile = blockIdx.x * 4 + wv;
  if (tile >= ntiles) return;
  const int row0 = tile * 16;
  const int lo = lane & 15, hi = lane >> 4;

  v16bf a0 = load_a_ea(ea + (size_t)(row0 + lo) * EDGE_IN, hi);
#pragma unroll
  for (int nt = 0; nt < 8; ++nt) {
    v8f acc = {};
    acc = wmma_bf16(a0, load_b_staged(w0s, 0, nt, lane), acc);
    float bias = b0[nt * 16 + lo];
#pragma unroll
    for (int r = 0; r < 8; ++r) {
      float v = acc[r] + bias;
      lds[wv][r + 8 * hi][nt * 16 + lo] = (__bf16)(v > 0.f ? v : 0.f);
    }
  }
  v16bf a[4];
#pragma unroll
  for (int kk = 0; kk < 4; ++kk)
    a[kk] = load_a_bf16_row((const __bf16*)&lds[wv][lo][0], hi, kk * 32);
#pragma unroll
  for (int nt = 0; nt < 8; ++nt) {
    v8f acc = {};
#pragma unroll
    for (int kk = 0; kk < 4; ++kk)
      acc = wmma_bf16(a[kk], load_b_staged(w1s, kk, nt, lane), acc);
    float bias = b1[nt * 16 + lo];
#pragma unroll
    for (int r = 0; r < 8; ++r)
      e_out[(size_t)(row0 + r + 8 * hi) * HID + nt * 16 + lo] = (__bf16)(acc[r] + bias);
  }
}

// fused: e_proj = e@We+be ; msg = relu(x[src]+e_proj) ; aggr[dst] += msg (L2-resident)
__global__ __launch_bounds__(128) void k_edge_msg(
    const __bf16* __restrict__ e, const __bf16* __restrict__ wes,
    const float* __restrict__ be128,  // padded to 128
    const int* __restrict__ src, const int* __restrict__ dst,
    const float* __restrict__ xin, float* __restrict__ aggr, int ntiles) {
  const int lane = threadIdx.x & 31;
  const int wv = threadIdx.x >> 5;
  const int tile = blockIdx.x * 4 + wv;
  if (tile >= ntiles) return;
  const int e0 = tile * 16;
  const int lo = lane & 15, hi = lane >> 4;

  v16bf a[4];
#pragma unroll
  for (int kk = 0; kk < 4; ++kk)
    a[kk] = load_a_bf16_row(e + (size_t)(e0 + lo) * HID, hi, kk * 32);

  int sidx[8], didx[8];
#pragma unroll
  for (int r = 0; r < 8; ++r) {
    int m = e0 + r + 8 * hi;
    sidx[r] = src[m];
    didx[r] = dst[m];
  }

#pragma unroll
  for (int nt = 0; nt < 8; ++nt) {
    v8f acc = {};
#pragma unroll
    for (int kk = 0; kk < 4; ++kk)
      acc = wmma_bf16(a[kk], load_b_staged(wes, kk, nt, lane), acc);
    const int col = nt * 16 + lo;
    const float bias = be128[col];
#pragma unroll
    for (int r = 0; r < 8; ++r) {
      float msg = acc[r] + bias + xin[(size_t)sidx[r] * HID + col];
      msg = msg > 0.f ? msg : 0.f;
      unsafeAtomicAdd(&aggr[(size_t)didx[r] * HID + col], msg);
    }
  }
}

// x_out = bn(relu(W2 @ relu(W1 @ (x+aggr) + b1) + b2))
__global__ __launch_bounds__(128) void k_node_mlp(
    const float* __restrict__ xin, const float* __restrict__ aggr,
    const __bf16* __restrict__ w1s, const float* __restrict__ b1,
    const __bf16* __restrict__ w2s, const float* __restrict__ b2,
    const float* __restrict__ g, const float* __restrict__ bt,
    const float* __restrict__ mu, const float* __restrict__ var,
    float* __restrict__ xout, int ntiles) {
  __shared__ __bf16 lds[4][16][160];
  const int lane = threadIdx.x & 31;
  const int wv = threadIdx.x >> 5;
  const int tile = blockIdx.x * 4 + wv;
  if (tile >= ntiles) return;
  const int row0 = tile * 16;
  const int lo = lane & 15, hi = lane >> 4;

  v16bf a[4];
#pragma unroll
  for (int kk = 0; kk < 4; ++kk)
    a[kk] = load_a_sum(xin + (size_t)(row0 + lo) * HID, aggr + (size_t)(row0 + lo) * HID,
                       hi, kk * 32);

#pragma unroll
  for (int nt = 0; nt < 8; ++nt) {
    v8f acc = {};
#pragma unroll
    for (int kk = 0; kk < 4; ++kk)
      acc = wmma_bf16(a[kk], load_b_staged(w1s, kk, nt, lane), acc);
    float bias = b1[nt * 16 + lo];
#pragma unroll
    for (int r = 0; r < 8; ++r) {
      float v = acc[r] + bias;
      lds[wv][r + 8 * hi][nt * 16 + lo] = (__bf16)(v > 0.f ? v : 0.f);
    }
  }

  v16bf a2[4];
#pragma unroll
  for (int kk = 0; kk < 4; ++kk)
    a2[kk] = load_a_bf16_row((const __bf16*)&lds[wv][lo][0], hi, kk * 32);

#pragma unroll
  for (int nt = 0; nt < 8; ++nt) {
    v8f acc = {};
#pragma unroll
    for (int kk = 0; kk < 4; ++kk)
      acc = wmma_bf16(a2[kk], load_b_staged(w2s, kk, nt, lane), acc);
    const int col = nt * 16 + lo;
    const float bias = b2[col];
    const float sc = g[col] * rsqrtf(var[col] + BN_EPS);
    const float sh = bt[col] - mu[col] * sc;
#pragma unroll
    for (int r = 0; r < 8; ++r) {
      float h = acc[r] + bias;
      h = h > 0.f ? h : 0.f;
      xout[(size_t)(row0 + r + 8 * hi) * HID + col] = sc * h + sh;
    }
  }
}

__global__ void k_pool_scatter(const float* __restrict__ x, const int* __restrict__ batch,
                               float* __restrict__ pool, float* __restrict__ cnt, int nN) {
  int i = blockIdx.x * blockDim.x + threadIdx.x;
  if (i >= nN * HID) return;
  int n = i >> 7, c = i & 127;
  int b = batch[n];
  unsafeAtomicAdd(&pool[(size_t)b * HID + c], x[i]);
  if (c == 0) unsafeAtomicAdd(&cnt[b], 1.f);
}

__global__ void k_pool_div(float* __restrict__ pool, const float* __restrict__ cnt, int nG) {
  int i = blockIdx.x * blockDim.x + threadIdx.x;
  if (i >= nG * HID) return;
  float c = cnt[i >> 7];
  pool[i] /= (c > 1.f ? c : 1.f);
}

// lat = W2 @ bn(relu(W1 @ lattice + b1)) + b2
__global__ __launch_bounds__(128) void k_lattice(
    const float* __restrict__ li, const __bf16* __restrict__ w1s, const float* __restrict__ b1,
    const float* __restrict__ g, const float* __restrict__ bt,
    const float* __restrict__ mu, const float* __restrict__ var,
    const __bf16* __restrict__ w2s, const float* __restrict__ b2,
    float* __restrict__ lo_out, int ntiles) {
  __shared__ __bf16 lds[4][16][160];
  const int lane = threadIdx.x & 31;
  const int wv = threadIdx.x >> 5;
  const int tile = blockIdx.x * 4 + wv;
  if (tile >= ntiles) return;
  const int row0 = tile * 16;
  const int lo = lane & 15, hi = lane >> 4;

  v16bf a0 = load_a_lat(li + (size_t)(row0 + lo) * LAT_IN, hi);
#pragma unroll
  for (int nt = 0; nt < 8; ++nt) {
    v8f acc = {};
    acc = wmma_bf16(a0, load_b_staged(w1s, 0, nt, lane), acc);
    const int col = nt * 16 + lo;
    const float bias = b1[col];
    const float sc = g[col] * rsqrtf(var[col] + BN_EPS);
    const float sh = bt[col] - mu[col] * sc;
#pragma unroll
    for (int r = 0; r < 8; ++r) {
      float h = acc[r] + bias;
      h = h > 0.f ? h : 0.f;
      lds[wv][r + 8 * hi][col] = (__bf16)(sc * h + sh);
    }
  }
  v16bf a[4];
#pragma unroll
  for (int kk = 0; kk < 4; ++kk)
    a[kk] = load_a_bf16_row((const __bf16*)&lds[wv][lo][0], hi, kk * 32);
#pragma unroll
  for (int nt = 0; nt < 8; ++nt) {
    v8f acc = {};
#pragma unroll
    for (int kk = 0; kk < 4; ++kk)
      acc = wmma_bf16(a[kk], load_b_staged(w2s, kk, nt, lane), acc);
    float bias = b2[nt * 16 + lo];
#pragma unroll
    for (int r = 0; r < 8; ++r)
      lo_out[(size_t)(row0 + r + 8 * hi) * HID + nt * 16 + lo] = acc[r] + bias;
  }
}

// out = W2 @ bn(relu(W1 @ [pool|lat] + b1)) + b2
__global__ __launch_bounds__(128) void k_final(
    const float* __restrict__ pool, const float* __restrict__ lat,
    const __bf16* __restrict__ w1s, const float* __restrict__ b1,
    const float* __restrict__ g, const float* __restrict__ bt,
    const float* __restrict__ mu, const float* __restrict__ var,
    const float* __restrict__ w2, const float* __restrict__ b2,
    float* __restrict__ out, int ntiles) {
  __shared__ __bf16 lds[4][16][160];
  const int lane = threadIdx.x & 31;
  const int wv = threadIdx.x >> 5;
  const int tile = blockIdx.x * 4 + wv;
  if (tile >= ntiles) return;
  const int row0 = tile * 16;
  const int lo = lane & 15, hi = lane >> 4;

  const float* prow = pool + (size_t)(row0 + lo) * HID;
  const float* lrow = lat + (size_t)(row0 + lo) * HID;
  v16bf a[8];
#pragma unroll
  for (int kk = 0; kk < 8; ++kk)  // kk<4 -> pool, kk>=4 -> lat (compile-time select)
    a[kk] = (kk < 4) ? load_a_f32_row(prow, hi, kk * 32)
                     : load_a_f32_row(lrow, hi, (kk - 4) * 32);

#pragma unroll
  for (int nt = 0; nt < 8; ++nt) {
    v8f acc = {};
#pragma unroll
    for (int kk = 0; kk < 8; ++kk)
      acc = wmma_bf16(a[kk], load_b_staged(w1s, kk, nt, lane), acc);
    const int col = nt * 16 + lo;
    const float bias = b1[col];
    const float sc = g[col] * rsqrtf(var[col] + BN_EPS);
    const float sh = bt[col] - mu[col] * sc;
#pragma unroll
    for (int r = 0; r < 8; ++r) {
      float h = acc[r] + bias;
      h = h > 0.f ? h : 0.f;
      lds[wv][r + 8 * hi][col] = (__bf16)(sc * h + sh);
    }
  }
  if (hi == 0) {
    float s = 0.f;
#pragma unroll
    for (int k = 0; k < HID; ++k) s += (float)lds[wv][lo][k] * w2[k];
    out[row0 + lo] = s + b2[0];
  }
}

// ================= host =================

extern "C" void kernel_launch(void* const* d_in, const int* in_sizes, int n_in,
                              void* d_out, int out_size, void* d_ws, size_t ws_size,
                              hipStream_t stream) {
  const int N = 50000, E = 1600000, G = 256;
  auto f = [&](int i) { return (const float*)d_in[i]; };

  const float* x       = f(0);
  const int*   ei      = (const int*)d_in[1];
  const float* ea      = f(2);
  const int*   batch   = (const int*)d_in[3];
  const float* lattice = f(4);

  const float *ee_w0 = f(5), *ee_b0 = f(6), *ee_w1 = f(7), *ee_b1 = f(8);
  struct Layer { const float *ew, *eb, *m1w, *m1b, *m2w, *m2b, *bng, *bnb, *bnm, *bnv; int in_dim; };
  Layer L[4];
  int q = 9;
  for (int i = 0; i < 4; ++i) {
    L[i] = { f(q), f(q+1), f(q+2), f(q+3), f(q+4), f(q+5), f(q+6), f(q+7), f(q+8), f(q+9),
             (i == 0) ? NODE_IN : HID };
    q += 10;
  }
  const float *lat_w1 = f(49), *lat_b1 = f(50), *lat_g = f(51), *lat_bt = f(52),
              *lat_m = f(53), *lat_v = f(54), *lat_w2 = f(55), *lat_b2 = f(56);
  const float *fin_w1 = f(57), *fin_b1 = f(58), *fin_g = f(59), *fin_bt = f(60),
              *fin_m = f(61), *fin_v = f(62), *fin_w2 = f(63), *fin_b2 = f(64);

  // ---- workspace carve-out ----
  char* ws = (char*)d_ws;
  auto carve = [&](size_t bytes) { char* r = ws; ws += (bytes + 255) & ~(size_t)255; return r; };
  // staged bf16 weights (WMMA-B layout, padded)
  __bf16* st_ee0  = (__bf16*)carve((size_t)32 * 128 * 2);
  __bf16* st_ee1  = (__bf16*)carve((size_t)128 * 128 * 2);
  __bf16* st_ew[4]; __bf16* st_m1[4]; __bf16* st_m2[4];
  for (int i = 0; i < 4; ++i) {
    st_ew[i] = (__bf16*)carve((size_t)128 * 128 * 2);
    st_m1[i] = (__bf16*)carve((size_t)128 * 128 * 2);
    st_m2[i] = (__bf16*)carve((size_t)128 * 128 * 2);
  }
  __bf16* st_lat1 = (__bf16*)carve((size_t)32 * 128 * 2);
  __bf16* st_lat2 = (__bf16*)carve((size_t)128 * 128 * 2);
  __bf16* st_fin1 = (__bf16*)carve((size_t)256 * 128 * 2);
  float*  eb0_pad = (float*)carve(HID * sizeof(float));
  // big buffers
  __bf16* e_bf = (__bf16*)carve((size_t)E * HID * sizeof(__bf16));
  float*  xA   = (float*)carve((size_t)N * HID * sizeof(float));
  float*  xB   = (float*)carve((size_t)N * HID * sizeof(float));
  float*  aggr = (float*)carve((size_t)N * HID * sizeof(float));
  float*  pool = (float*)carve((size_t)G * HID * sizeof(float));
  float*  cnt  = (float*)carve((size_t)G * sizeof(float));
  float*  lat  = (float*)carve((size_t)G * HID * sizeof(float));

  auto stage = [&](const float* w, __bf16* outp, int K, int Nc, int KP) {
    int total = KP * 128;
    k_stage_b<<<(total + 255) / 256, 256, 0, stream>>>(w, outp, K, Nc, KP, 128);
  };
  stage(ee_w0, st_ee0, EDGE_IN, HID, 32);
  stage(ee_w1, st_ee1, HID, HID, 128);
  for (int i = 0; i < 4; ++i) {
    stage(L[i].ew,  st_ew[i], HID, L[i].in_dim, 128);
    stage(L[i].m1w, st_m1[i], L[i].in_dim, HID, 128);
    stage(L[i].m2w, st_m2[i], HID, HID, 128);
  }
  stage(lat_w1, st_lat1, LAT_IN, HID, 32);
  stage(lat_w2, st_lat2, HID, HID, 128);
  stage(fin_w1, st_fin1, 2 * HID, HID, 256);
  k_pad_vec<<<1, 128, 0, stream>>>(L[0].eb, eb0_pad, NODE_IN);

  const int* srcp = ei;
  const int* dstp = ei + E;

  k_pad_x<<<(N * HID + 255) / 256, 256, 0, stream>>>(x, xA, N * HID);
  k_edge_encoder<<<E / 64, 128, 0, stream>>>(ea, st_ee0, ee_b0, st_ee1, ee_b1, e_bf, E / 16);

  float* xin = xA;
  float* xout = xB;
  const int ntilesN = (N + 15) / 16;  // 3125
  for (int i = 0; i < 4; ++i) {
    k_zero<<<(N * HID + 255) / 256, 256, 0, stream>>>(aggr, N * HID);
    const float* eb = (i == 0) ? eb0_pad : L[i].eb;
    k_edge_msg<<<E / 64, 128, 0, stream>>>(e_bf, st_ew[i], eb, srcp, dstp, xin, aggr, E / 16);
    k_node_mlp<<<(ntilesN + 3) / 4, 128, 0, stream>>>(
        xin, aggr, st_m1[i], L[i].m1b, st_m2[i], L[i].m2b,
        L[i].bng, L[i].bnb, L[i].bnm, L[i].bnv, xout, ntilesN);
    float* t = xin; xin = xout; xout = t;
  }

  k_zero<<<(G * HID + 255) / 256, 256, 0, stream>>>(pool, G * HID);
  k_zero<<<1, 256, 0, stream>>>(cnt, G);
  k_pool_scatter<<<(N * HID + 255) / 256, 256, 0, stream>>>(xin, batch, pool, cnt, N);
  k_pool_div<<<(G * HID + 255) / 256, 256, 0, stream>>>(pool, cnt, G);

  k_lattice<<<4, 128, 0, stream>>>(lattice, st_lat1, lat_b1, lat_g, lat_bt, lat_m, lat_v,
                                   st_lat2, lat_b2, lat, G / 16);
  k_final<<<4, 128, 0, stream>>>(pool, lat, st_fin1, fin_b1, fin_g, fin_bt, fin_m, fin_v,
                                 fin_w2, fin_b2, (float*)d_out, G / 16);
}
